// VanillaRNN_20899310863144
// MI455X (gfx1250) — matrix operand order repaired and checked by reading, not compiled, not run on hardware
//
#include <hip/hip_runtime.h>

typedef __attribute__((ext_vector_type(16))) _Float16 v16h;
typedef __attribute__((ext_vector_type(8)))  _Float16 v8h;
typedef __attribute__((ext_vector_type(8)))  float    v8f;
typedef __attribute__((ext_vector_type(4)))  float    v4f;

#define RNN_H 128
#define RNN_T 512
#define RNN_C 10
#define HSTRIDE 136   // halves per column (128 + 8 pad -> conflict-free LDS)

__device__ __forceinline__ float fast_tanh(float x) {
    // tanh(x) = 1 - 2/(exp(2x)+1); exp(2x) = exp2(x * 2*log2(e))
    float e = __builtin_amdgcn_exp2f(x * 2.88539008177792681f);
    return 1.0f - 2.0f * __builtin_amdgcn_rcpf(e + 1.0f);
}

__global__ __launch_bounds__(32, 1)
void VanillaRNN_wmma_kernel(const float* __restrict__ x,
                            const float* __restrict__ Whx,
                            const float* __restrict__ Whh,
                            const float* __restrict__ Wph,
                            const float* __restrict__ bh,
                            const float* __restrict__ bp,
                            float* __restrict__ out) {
    // One wave32 per 16 batch columns. h tile (128 x 16, f16, column-major) in LDS.
    __shared__ _Float16 hbuf[16 * HSTRIDE];

    const int lane = threadIdx.x;      // 0..31
    const int n    = lane & 15;        // column within tile
    const int hi   = lane >> 4;        // lane-half selector
    const int b0   = blockIdx.x * 16;  // first batch column of this tile

    // ---- zero-init h0 (1088 dwords, 34 full-wave stores, no divergence) ----
    unsigned int* hz = (unsigned int*)hbuf;
    #pragma unroll
    for (int i = 0; i < 34; ++i) hz[lane + 32 * i] = 0u;

    // ---- W_hh resident in VGPRs, f16, WMMA A-layout ----
    // A-layout (16x32 f16): lane holds row m = 16*mt + (lane&15);
    // lanes 0-15: elems 0..7 = K 0..7,   elems 8..15 = K 16..23
    // lanes16-31: elems 0..7 = K 8..15,  elems 8..15 = K 24..31
    v16h A[4][8];
    #pragma unroll
    for (int kc = 0; kc < 4; ++kc) {
        #pragma unroll
        for (int mt = 0; mt < 8; ++mt) {
            const int m  = 16 * mt + n;
            const int kb = 32 * kc + 8 * hi;
            const float* wrow = Whh + m * RNN_H;
            v4f w0 = *(const v4f*)(wrow + kb);
            v4f w1 = *(const v4f*)(wrow + kb + 4);
            v4f w2 = *(const v4f*)(wrow + kb + 16);
            v4f w3 = *(const v4f*)(wrow + kb + 20);
            v16h a;
            #pragma unroll
            for (int i = 0; i < 4; ++i) {
                a[i]      = (_Float16)w0[i];
                a[4 + i]  = (_Float16)w1[i];
                a[8 + i]  = (_Float16)w2[i];
                a[12 + i] = (_Float16)w3[i];
            }
            A[kc][mt] = a;
        }
    }

    // ---- per-lane f32 bias and W_hx (accumulator-layout scalars) ----
    // C/D layout: VGPR r of tile mt holds row m = 16*mt + r + 8*hi, col n = lane&15
    float BH[8][8], WX[8][8];
    #pragma unroll
    for (int mt = 0; mt < 8; ++mt) {
        const int mb = 16 * mt + 8 * hi;
        v4f blo = *(const v4f*)(bh + mb);
        v4f bhi = *(const v4f*)(bh + mb + 4);
        v4f wlo = *(const v4f*)(Whx + mb);
        v4f whi = *(const v4f*)(Whx + mb + 4);
        #pragma unroll
        for (int r = 0; r < 4; ++r) {
            BH[mt][r]     = blo[r];
            BH[mt][r + 4] = bhi[r];
            WX[mt][r]     = wlo[r];
            WX[mt][r + 4] = whi[r];
        }
    }

    // ---- recurrence over time ----
    const float* xp = x + (b0 + n) * RNN_T;  // this lane's column of x
    for (int t = 0; t < RNN_T; ++t) {
        float xt = xp[t];

        // acc = W_hx * x_t (outer product) + b_h, exact f32
        v8f acc[8];
        #pragma unroll
        for (int mt = 0; mt < 8; ++mt) {
            #pragma unroll
            for (int r = 0; r < 8; ++r)
                acc[mt][r] = fmaf(WX[mt][r], xt, BH[mt][r]);
        }

        // B operand (32x16 f16): lanes 0-15 -> K rows kb..kb+7, kb+16..kb+23 of column n,
        // lanes 16-31 -> K rows kb+8..kb+15, kb+24..kb+31  (matches A layout)
        v16h Bv[4];
        #pragma unroll
        for (int kc = 0; kc < 4; ++kc) {
            const _Float16* col = hbuf + n * HSTRIDE + 32 * kc + 8 * hi;
            v8h lo  = *(const v8h*)(col);
            v8h hi8 = *(const v8h*)(col + 16);
            v16h b;
            #pragma unroll
            for (int i = 0; i < 8; ++i) { b[i] = lo[i]; b[8 + i] = hi8[i]; }
            Bv[kc] = b;
        }

        // acc += W_hh @ h   (32 x v_wmma_f32_16x16x32_f16, dep distance 8)
        #pragma unroll
        for (int kc = 0; kc < 4; ++kc) {
            #pragma unroll
            for (int mt = 0; mt < 8; ++mt) {
                acc[mt] = __builtin_amdgcn_wmma_f32_16x16x32_f16(
                    false, A[kc][mt], false, Bv[kc],
                    (short)0, acc[mt], false, false);
            }
        }

        // h = tanh(acc), pack f16 pairs, write back (rows m,m+1 contiguous)
        #pragma unroll
        for (int mt = 0; mt < 8; ++mt) {
            const int mb = 16 * mt + 8 * hi;
            #pragma unroll
            for (int r = 0; r < 8; r += 2) {
                float t0 = fast_tanh(acc[mt][r]);
                float t1 = fast_tanh(acc[mt][r + 1]);
                auto pk = __builtin_amdgcn_cvt_pkrtz(t0, t1);
                *reinterpret_cast<decltype(pk)*>(hbuf + n * HSTRIDE + mb + r) = pk;
            }
        }
    }

    // ---- final projection p = W_ph @ h_T + b_p ; out[b][c] = p[c][b] ----
    // lanes 0-15 handle even c, lanes 16-31 odd c; each lane: column n.
    #pragma unroll
    for (int j = 0; j < 5; ++j) {
        const int c = 2 * j + hi;
        float s = bp[c];
        const float* wr = Wph + c * RNN_H;
        const _Float16* col = hbuf + n * HSTRIDE;
        #pragma unroll 8
        for (int k = 0; k < RNN_H; ++k)
            s = fmaf(wr[k], (float)col[k], s);
        out[(b0 + n) * RNN_C + c] = s;
    }
}

extern "C" void kernel_launch(void* const* d_in, const int* in_sizes, int n_in,
                              void* d_out, int out_size, void* d_ws, size_t ws_size,
                              hipStream_t stream) {
    const float* x   = (const float*)d_in[0];
    const float* Whx = (const float*)d_in[1];
    const float* Whh = (const float*)d_in[2];
    const float* Wph = (const float*)d_in[3];
    const float* bh  = (const float*)d_in[4];
    const float* bp  = (const float*)d_in[5];
    float* out = (float*)d_out;

    const int B = in_sizes[0] / RNN_T;       // 4096
    dim3 grid(B / 16);                       // 256 independent waves
    dim3 block(32);
    hipLaunchKernelGGL(VanillaRNN_wmma_kernel, grid, block, 0, stream,
                       x, Whx, Whh, Wph, bh, bp, out);
}
